// DiscriminativeClueCorrection_29205777613506
// MI455X (gfx1250) — matrix operand order repaired and checked by reading, not compiled, not run on hardware
//
#include <hip/hip_runtime.h>

#define DEV __device__ __forceinline__

typedef __attribute__((ext_vector_type(16))) __bf16 bf16x16;
typedef __attribute__((ext_vector_type(8)))  __bf16 bf16x8;
typedef __attribute__((ext_vector_type(4)))  __bf16 bf16x4;
typedef __attribute__((ext_vector_type(8)))  float  f32x8;
typedef __attribute__((ext_vector_type(4)))  float  f32x4;

constexpr int Bd   = 512;
constexpr int Md   = 96;
constexpr int Dd   = 512;
constexpr int Hd   = 8;
constexpr int DHd  = 64;
constexpr int BM   = Bd * Md;     // 49152 rows of text
constexpr int NNEG = 1536;        // roll(corr) ++ corr ++ memory[512:1024]
constexpr float EPSn = 1e-8f;

DEV float waveSum(float v) {
#pragma unroll
  for (int m = 16; m >= 1; m >>= 1) v += __shfl_xor(v, m, 32);
  return v;
}

DEV bf16x16 cat8(bf16x8 lo, bf16x8 hi) {
  return __builtin_shufflevector(lo, hi, 0,1,2,3,4,5,6,7,8,9,10,11,12,13,14,15);
}

// ---------------------------------------------------------------------------
// K1: per-b: normalize vis (store vis_n), sims[b][m] = cos(vis, text[b,m])
// ---------------------------------------------------------------------------
__global__ void sims_kernel(const float* __restrict__ vis,
                            const float* __restrict__ text,
                            float* __restrict__ vis_n,
                            float* __restrict__ sims) {
  const int b = blockIdx.x;
  const int tid = threadIdx.x;
  const int wave = tid >> 5, lane = tid & 31;
  __shared__ float visn[Dd];
  __shared__ float red[8];
  float ss = 0.f;
  for (int i = tid; i < Dd; i += 256) { float x = vis[b*Dd + i]; visn[i] = x; ss += x*x; }
  ss = waveSum(ss);
  if (lane == 0) red[wave] = ss;
  __syncthreads();
  if (tid == 0) {
    float t = 0.f;
    for (int i = 0; i < 8; ++i) t += red[i];
    red[0] = 1.f / fmaxf(sqrtf(t), EPSn);
  }
  __syncthreads();
  const float inv = red[0];
  for (int i = tid; i < Dd; i += 256) { float x = visn[i]*inv; visn[i] = x; vis_n[b*Dd + i] = x; }
  __syncthreads();
  for (int m = wave; m < Md; m += 8) {
    const float* tp = text + ((size_t)b*Md + m) * Dd;
    float dot = 0.f, tss = 0.f;
    for (int d = lane; d < Dd; d += 32) { float t = tp[d]; dot += visn[d]*t; tss += t*t; }
    dot = waveSum(dot); tss = waveSum(tss);
    if (lane == 0) sims[b*Md + m] = dot / fmaxf(sqrtf(tss), EPSn);
  }
}

// ---------------------------------------------------------------------------
// K2: per-b: sort 96 sims descending (odd-even), group-of-32 mean/std(ddof=1)
// ---------------------------------------------------------------------------
__global__ void cluster_kernel(const float* __restrict__ sims,
                               float* __restrict__ scores) {
  const int b = blockIdx.x;
  const int tid = threadIdx.x;
  __shared__ float s[Md];
  if (tid < Md) s[tid] = sims[b*Md + tid];
  __syncthreads();
  for (int it = 0; it < Md; ++it) {
    int i = 2*tid + (it & 1);
    if (i + 1 < Md) {
      float a = s[i], c = s[i+1];
      if (a < c) { s[i] = c; s[i+1] = a; }
    }
    __syncthreads();
  }
  if (tid < 3) {
    float sum = 0.f, sq = 0.f;
    for (int j = 0; j < 32; ++j) { float x = s[tid*32 + j]; sum += x; sq += x*x; }
    float mean = sum / 32.f;
    float var  = fmaxf((sq - 32.f*mean*mean) / 31.f, 0.f);
    scores[b*3 + tid] = mean / (sqrtf(var) + 1e-6f);
  }
}

// ---------------------------------------------------------------------------
// K3: C = A(f32, rows x 512) @ W^T(f32, ncols x 512) + bias
//     Double-buffered LDS pipeline: float4 global loads -> packed bf16 cvt ->
//     b64 LDS stores; v_wmma_f32_16x16x32_bf16 accumulation.
//     Block tile 128x64, 8 waves, each wave 2x2 16x16 tiles. NW=2 fuses two
//     weight matrices over one A tile (K/V projection).
// ---------------------------------------------------------------------------
template <int NW, bool OUT_BF16>
__global__ __launch_bounds__(256)
void gemm_wmma(const float* __restrict__ A,
               const float* __restrict__ W0, const float* __restrict__ bias0, void* __restrict__ out0,
               const float* __restrict__ W1, const float* __restrict__ bias1, void* __restrict__ out1,
               int ncols) {
  const int tid  = threadIdx.x;
  const int lane = tid & 31;
  const int wid  = tid >> 5;
  const int wm   = wid & 3;           // 4 row strips of 32
  const int wn   = wid >> 2;          // 2 col strips of 32
  const int l16  = lane & 15;
  const int hi   = lane >> 4;         // half-wave select
  const int rowBase = blockIdx.x * 128;
  const int colBase = blockIdx.y * 64;

  __shared__ __align__(16) __bf16 As[2][128][40];
  __shared__ __align__(16) __bf16 Ws[2][NW][64][40];

  f32x8 acc[NW][2][2];
  const f32x8 z = {0.f,0.f,0.f,0.f,0.f,0.f,0.f,0.f};
#pragma unroll
  for (int w = 0; w < NW; ++w)
#pragma unroll
    for (int i = 0; i < 2; ++i)
#pragma unroll
      for (int j = 0; j < 2; ++j) acc[w][i][j] = z;

  // staging registers: A tile 128x32 = 4 float4/thread, each W tile = 2/thread
  f32x4 areg[4];
  f32x4 wreg[NW][2];

  auto loadTile = [&](int kk) {
#pragma unroll
    for (int i = 0; i < 4; ++i) {
      int e = tid + i*256;            // float4 slot
      int r = e >> 3, c = (e & 7) * 4;
      areg[i] = *reinterpret_cast<const f32x4*>(A + (size_t)(rowBase + r)*Dd + kk + c);
    }
#pragma unroll
    for (int w = 0; w < NW; ++w) {
      const float* Wp = (w == 0) ? W0 : W1;
#pragma unroll
      for (int i = 0; i < 2; ++i) {
        int e = tid + i*256;
        int r = e >> 3, c = (e & 7) * 4;
        wreg[w][i] = *reinterpret_cast<const f32x4*>(Wp + (size_t)(colBase + r)*Dd + kk + c);
      }
    }
  };
  auto storeTile = [&](int buf) {
#pragma unroll
    for (int i = 0; i < 4; ++i) {
      int e = tid + i*256;
      int r = e >> 3, c = (e & 7) * 4;
      bf16x4 bv = { (__bf16)areg[i][0], (__bf16)areg[i][1],
                    (__bf16)areg[i][2], (__bf16)areg[i][3] };
      *reinterpret_cast<bf16x4*>(&As[buf][r][c]) = bv;
    }
#pragma unroll
    for (int w = 0; w < NW; ++w) {
#pragma unroll
      for (int i = 0; i < 2; ++i) {
        int e = tid + i*256;
        int r = e >> 3, c = (e & 7) * 4;
        bf16x4 bv = { (__bf16)wreg[w][i][0], (__bf16)wreg[w][i][1],
                      (__bf16)wreg[w][i][2], (__bf16)wreg[w][i][3] };
        *reinterpret_cast<bf16x4*>(&Ws[buf][w][r][c]) = bv;
      }
    }
  };

  constexpr int KSTEPS = Dd / 32;     // 16
  loadTile(0);
  storeTile(0);
  for (int ki = 0; ki < KSTEPS; ++ki) {
    __syncthreads();                  // tile (ki&1) visible; prior reads done
    if (ki + 1 < KSTEPS) loadTile((ki + 1) * 32);   // overlap with WMMA below

    const int buf = ki & 1;
    // A frag: lanes 0-15 -> K {0..7,16..23}, lanes 16-31 -> K {8..15,24..31}
    bf16x16 afrag[2];
#pragma unroll
    for (int fa = 0; fa < 2; ++fa) {
      int r = wm*32 + fa*16 + l16;
      bf16x8 lo  = *reinterpret_cast<const bf16x8*>(&As[buf][r][hi*8]);
      bf16x8 hiv = *reinterpret_cast<const bf16x8*>(&As[buf][r][16 + hi*8]);
      afrag[fa] = cat8(lo, hiv);
    }
    // B frag: lane holds N=l16, K {hi*8..hi*8+7, 16+hi*8..16+hi*8+7}
#pragma unroll
    for (int w = 0; w < NW; ++w) {
#pragma unroll
      for (int fb = 0; fb < 2; ++fb) {
        int ccol = wn*32 + fb*16 + l16;
        bf16x8 lo  = *reinterpret_cast<const bf16x8*>(&Ws[buf][w][ccol][hi*8]);
        bf16x8 hiv = *reinterpret_cast<const bf16x8*>(&Ws[buf][w][ccol][16 + hi*8]);
        bf16x16 bfrag = cat8(lo, hiv);
#pragma unroll
        for (int fa = 0; fa < 2; ++fa) {
          acc[w][fa][fb] = __builtin_amdgcn_wmma_f32_16x16x32_bf16(
              false, afrag[fa], false, bfrag, (short)0, acc[w][fa][fb], false, false);
        }
      }
    }
    if (ki + 1 < KSTEPS) storeTile((ki + 1) & 1);   // cvt + ds_store next tile
  }

  // store: C VGPR r -> M = fa*16 + hi*8 + r, N = fb*16 + l16
#pragma unroll
  for (int w = 0; w < NW; ++w) {
    const float* bp = (w == 0) ? bias0 : bias1;
    void* op = (w == 0) ? out0 : out1;
#pragma unroll
    for (int fb = 0; fb < 2; ++fb) {
      int n = wn*32 + fb*16 + l16;
      float bv = bp ? bp[colBase + n] : 0.f;
#pragma unroll
      for (int fa = 0; fa < 2; ++fa) {
#pragma unroll
        for (int vr = 0; vr < 8; ++vr) {
          int m = wm*32 + fa*16 + hi*8 + vr;
          float v = acc[w][fa][fb][vr] + bv;
          size_t idx = (size_t)(rowBase + m) * (size_t)ncols + (size_t)(colBase + n);
          if (OUT_BF16) ((__bf16*)op)[idx] = (__bf16)v;
          else          ((float*)op)[idx] = v;
        }
      }
    }
  }
}

// ---------------------------------------------------------------------------
// K4: per (b,h): scores = q.k/8, softmax over 96 keys, ctx = attn.v  (f32)
// ---------------------------------------------------------------------------
__global__ void attn_kernel(const float* __restrict__ q,
                            const __bf16* __restrict__ kmat,
                            const __bf16* __restrict__ vmat,
                            float* __restrict__ ctx) {
  const int bh = blockIdx.x;
  const int b = bh >> 3, h = bh & 7;
  const int tid = threadIdx.x;
  __shared__ float qs[DHd];
  __shared__ float sc[Md];
  __shared__ float stat;
  if (tid < DHd) qs[tid] = q[b*Dd + h*DHd + tid];
  __syncthreads();
  if (tid < Md) {
    const bf16x8* kp8 = reinterpret_cast<const bf16x8*>(
        kmat + ((size_t)b*Md + tid)*Dd + h*DHd);   // 128B-aligned
    float s = 0.f;
#pragma unroll
    for (int j = 0; j < DHd/8; ++j) {
      bf16x8 kv = kp8[j];
#pragma unroll
      for (int t = 0; t < 8; ++t) s += qs[j*8 + t] * (float)kv[t];
    }
    sc[tid] = s * 0.125f;               // 1/sqrt(64)
  }
  __syncthreads();
  if (tid == 0) {
    float mx = sc[0];
    for (int m = 1; m < Md; ++m) mx = fmaxf(mx, sc[m]);
    stat = mx;
  }
  __syncthreads();
  if (tid < Md) sc[tid] = expf(sc[tid] - stat);
  __syncthreads();
  if (tid == 0) {
    float sm = 0.f;
    for (int m = 0; m < Md; ++m) sm += sc[m];
    stat = 1.f / sm;
  }
  __syncthreads();
  if (tid < Md) sc[tid] *= stat;
  __syncthreads();
  if (tid < DHd) {
    const __bf16* vp = vmat + (size_t)b*Md*Dd + h*DHd + tid;
    float a = 0.f;
    for (int m = 0; m < Md; ++m) a += sc[m] * (float)vp[(size_t)m*Dd];
    ctx[b*Dd + h*DHd + tid] = a;
  }
}

// ---------------------------------------------------------------------------
// K5: build normalized all_negs rows (f32): [roll(corr,1); corr; memory[512:]]
// ---------------------------------------------------------------------------
__global__ void negs_norm_kernel(const float* __restrict__ corrected,
                                 const float* __restrict__ text_memory,
                                 float* __restrict__ negs_n) {
  const int r = blockIdx.x;
  const int tid = threadIdx.x;
  const int wave = tid >> 5, lane = tid & 31;
  const float* src;
  if (r < Bd)          src = corrected   + (size_t)((r + Bd - 1) & (Bd - 1)) * Dd;
  else if (r < 2*Bd)   src = corrected   + (size_t)(r - Bd) * Dd;
  else                 src = text_memory + (size_t)(r - Bd) * Dd;  // rows 512..1023
  __shared__ float red[8];
  float ss = 0.f;
  for (int i = tid; i < Dd; i += 256) { float x = src[i]; ss += x*x; }
  ss = waveSum(ss);
  if (lane == 0) red[wave] = ss;
  __syncthreads();
  if (tid == 0) {
    float t = 0.f;
    for (int i = 0; i < 8; ++i) t += red[i];
    red[0] = 1.f / fmaxf(sqrtf(t), EPSn);
  }
  __syncthreads();
  const float inv = red[0];
  for (int i = tid; i < Dd; i += 256) negs_n[(size_t)r*Dd + i] = src[i] * inv;
}

// ---------------------------------------------------------------------------
// K6: per-b top-5 of 1536 sims; pos_sim = neg_sims[b][512+b] (self column)
// ---------------------------------------------------------------------------
__global__ void topk_loss_kernel(const float* __restrict__ neg_sims,
                                 const float* __restrict__ tau_p_log,
                                 const float* __restrict__ tau_n_log,
                                 float* __restrict__ partial) {
  int b = blockIdx.x * blockDim.x + threadIdx.x;
  if (b >= Bd) return;
  const float* row = neg_sims + (size_t)b * NNEG;
  float t[5] = {-1e30f,-1e30f,-1e30f,-1e30f,-1e30f};
  for (int j = 0; j < NNEG; ++j) {
    float v = row[j];
    if (v > t[4]) {
      int k = 4;
      while (k > 0 && t[k-1] < v) { t[k] = t[k-1]; --k; }
      t[k] = v;
    }
  }
  float tau_p = expf(tau_p_log[0]);
  float tau_n = expf(tau_n_log[0]);
  float neg = 0.f;
  for (int i = 0; i < 5; ++i) neg += expf(t[i] / tau_n);
  float p = expf(row[Bd + b] / tau_p);
  partial[b] = -logf(p / (p + neg + 1e-8f));
}

__global__ void finalize_loss_kernel(const float* __restrict__ partial,
                                     float* __restrict__ out) {
  const int tid = threadIdx.x;
  const int wave = tid >> 5, lane = tid & 31;
  __shared__ float red[8];
  float s = partial[tid] + partial[tid + 256];
  s = waveSum(s);
  if (lane == 0) red[wave] = s;
  __syncthreads();
  if (tid == 0) {
    float t = 0.f;
    for (int i = 0; i < 8; ++i) t += red[i];
    out[0] = t / (float)Bd;
  }
}

// ---------------------------------------------------------------------------
extern "C" void kernel_launch(void* const* d_in, const int* in_sizes, int n_in,
                              void* d_out, int out_size, void* d_ws, size_t ws_size,
                              hipStream_t stream) {
  (void)in_sizes; (void)n_in; (void)out_size; (void)ws_size;
  const float* vis   = (const float*)d_in[0];
  const float* text  = (const float*)d_in[1];
  const float* in_w  = (const float*)d_in[2];
  const float* in_b  = (const float*)d_in[3];
  const float* out_w = (const float*)d_in[4];
  const float* out_b = (const float*)d_in[5];
  const float* mem   = (const float*)d_in[6];
  const float* taupl = (const float*)d_in[7];
  const float* taunl = (const float*)d_in[8];

  float* outf      = (float*)d_out;
  float* loss      = outf;                 // [1]
  float* corrected = outf + 1;             // [512*512]
  float* scores    = outf + 1 + Bd*Dd;     // [512*3]

  char* ws = (char*)d_ws;
  size_t off = 0;
  auto carve = [&](size_t bytes) -> void* {
    off = (off + 255) & ~(size_t)255;
    void* p = ws + off;
    off += bytes;
    return p;
  };
  float*  q_f32    = (float*)carve((size_t)Bd*Dd*4);
  float*  sims     = (float*)carve((size_t)Bd*Md*4);
  float*  vis_n    = (float*)carve((size_t)Bd*Dd*4);
  float*  ctxbuf   = (float*)carve((size_t)Bd*Dd*4);
  float*  negs_n   = (float*)carve((size_t)NNEG*Dd*4);
  float*  neg_sims = (float*)carve((size_t)Bd*NNEG*4);
  float*  partial  = (float*)carve((size_t)Bd*4);
  __bf16* k_bf     = (__bf16*)carve((size_t)BM*Dd*2);
  __bf16* v_bf     = (__bf16*)carve((size_t)BM*Dd*2);

  const float* wq = in_w;
  const float* wk = in_w + (size_t)Dd*Dd;
  const float* wv = in_w + (size_t)2*Dd*Dd;
  const float* bq = in_b;
  const float* bk = in_b + Dd;
  const float* bv = in_b + 2*Dd;

  sims_kernel<<<Bd, 256, 0, stream>>>(vis, text, vis_n, sims);
  cluster_kernel<<<Bd, 128, 0, stream>>>(sims, scores);
  // q = vis @ Wq^T + bq
  gemm_wmma<1,false><<<dim3(Bd/128, Dd/64), 256, 0, stream>>>(
      vis, wq, bq, q_f32, nullptr, nullptr, nullptr, Dd);
  // k,v = text @ {Wk,Wv}^T + b  (fused, bf16 outputs)
  gemm_wmma<2,true ><<<dim3(BM/128, Dd/64), 256, 0, stream>>>(
      text, wk, bk, (void*)k_bf, wv, bv, (void*)v_bf, Dd);
  attn_kernel<<<Bd*Hd, 128, 0, stream>>>(q_f32, k_bf, v_bf, ctxbuf);
  // corrected = ctx @ Wout^T + bout
  gemm_wmma<1,false><<<dim3(Bd/128, Dd/64), 256, 0, stream>>>(
      ctxbuf, out_w, out_b, corrected, nullptr, nullptr, nullptr, Dd);
  negs_norm_kernel<<<NNEG, 256, 0, stream>>>(corrected, mem, negs_n);
  // neg_sims = vis_n @ negs_n^T
  gemm_wmma<1,false><<<dim3(Bd/128, NNEG/64), 256, 0, stream>>>(
      vis_n, negs_n, nullptr, neg_sims, nullptr, nullptr, nullptr, NNEG);
  topk_loss_kernel<<<2, 256, 0, stream>>>(neg_sims, taupl, taunl, partial);
  finalize_loss_kernel<<<1, 256, 0, stream>>>(partial, loss);
}
